// GRU_63316407878386
// MI455X (gfx1250) — matrix operand order repaired and checked by reading, not compiled, not run on hardware
//
#include <hip/hip_runtime.h>

typedef __bf16 bf16_t;
typedef __attribute__((ext_vector_type(16))) __bf16    v16bf;
typedef __attribute__((ext_vector_type(8)))  float     v8f;
typedef __attribute__((ext_vector_type(4)))  unsigned  uv4;

#define L_SEQ 512
#define NB    64
#define HD    512
#define ED    256
#define KCH   24      // 768 / 32 K-chunks (16 h-chunks + 8 e-chunks)
#define GWG   16      // persistent workgroups, 32 H-columns each

// workspace byte offsets (all 16B aligned)
#define WS_CNT   0            // 1024 * 4B barrier counters
#define WS_HXH   4096         // 64*512 bf16  (h part of hx, bf16)
#define WS_RH    69632        // 64*512 bf16  (r*h part of rhx, bf16)
#define WS_WSWZ  135168       // 3*768*512 bf16 swizzled weights (2.25 MB)
#define WS_XE    2494464      // 512*64*256 bf16 embedded inputs (16.8 MB)

struct U8x32 { uv4 a, b; };   // 32B carrier for one lane's 16 bf16 fragment slice
struct H8    { bf16_t h[8]; };

// fast device math: v_tanh_f32 / v_rcp_f32 / v_exp_f32 paths
__device__ inline float fast_tanh(float x) {
#if __has_builtin(__builtin_amdgcn_tanhf)
  return __builtin_amdgcn_tanhf(x);
#else
  float e2 = __expf(2.0f * x);
  return 1.0f - 2.0f * __builtin_amdgcn_rcpf(e2 + 1.0f);
#endif
}
__device__ inline float fast_sigmoid(float x) {
#if __has_builtin(__builtin_amdgcn_tanhf)
  return fmaf(0.5f, __builtin_amdgcn_tanhf(0.5f * x), 0.5f);
#else
  return __builtin_amdgcn_rcpf(1.0f + __expf(-x));
#endif
}

// ---------------------------------------------------------------- init ------
__global__ void gru_init(unsigned* __restrict__ cnt, unsigned* __restrict__ hxh_u32) {
  int tid = blockIdx.x * blockDim.x + threadIdx.x;
  if (tid < 1024)  cnt[tid] = 0u;
  if (tid < 16384) hxh_u32[tid] = 0u;   // h0 = 0 (bf16 pairs)
}

// ------------------------------------------------------- embedding gather ---
__global__ void gru_embed(const int* __restrict__ x, const float* __restrict__ emb,
                          bf16_t* __restrict__ xe) {
  int tid  = blockIdx.x * blockDim.x + threadIdx.x;  // L*NB*ED/8 = 1048576
  int row  = tid >> 5;                               // (t, n) flat row
  int e0   = (tid & 31) << 3;
  int idx  = x[row];
  const float* src = emb + (size_t)idx * ED + e0;
  H8 o;
#pragma unroll
  for (int i = 0; i < 8; ++i) o.h[i] = (bf16_t)src[i];
  *(uv4*)(xe + (size_t)row * ED + e0) = __builtin_bit_cast(uv4, o);
}

// --------------------------------------- weight swizzle -> WMMA B fragments -
// Layout: [gate(3)][colTile(32)][kchunk(24)][lane(32)][16 bf16]
// B 32x16 bf16 k order used here (same as A): lane<16 -> K = 32c..32c+15 ; lane>=16 -> K = 32c+16..31
// elem i -> K = kBase + i ; N = colTile*16 + (lane&15)
__global__ void gru_swizzle(const float* __restrict__ Wz, const float* __restrict__ Wr,
                            const float* __restrict__ Wh, bf16_t* __restrict__ Wswz) {
  int tid   = blockIdx.x * blockDim.x + threadIdx.x;  // 73728
  int lane  = tid & 31;
  int fr    = tid >> 5;                               // 0..2303
  int c     = fr % KCH;
  int t2    = fr / KCH;
  int tileG = t2 & 31;
  int g     = t2 >> 5;
  const float* W = (g == 0) ? Wz : (g == 1) ? Wr : Wh;
  int colIdx = tileG * 16 + (lane & 15);
  int kBase  = c * 32 + ((lane >= 16) ? 16 : 0);
  H8 o0, o1;
#pragma unroll
  for (int i = 0; i < 8; ++i) o0.h[i] = (bf16_t)W[(size_t)(kBase + i) * HD + colIdx];
#pragma unroll
  for (int i = 0; i < 8; ++i) o1.h[i] = (bf16_t)W[(size_t)(kBase + 8 + i) * HD + colIdx];
  uv4* dst = (uv4*)(Wswz + (size_t)fr * 512 + lane * 16);
  dst[0] = __builtin_bit_cast(uv4, o0);
  dst[1] = __builtin_bit_cast(uv4, o1);
}

// ---------------------------------------------------------------- barrier ---
__device__ inline void grid_barrier(unsigned* __restrict__ c, unsigned target) {
  __threadfence();            // make my stores device-visible
  __syncthreads();
  if (threadIdx.x == 0) {
    __hip_atomic_fetch_add(c, 1u, __ATOMIC_RELEASE, __HIP_MEMORY_SCOPE_AGENT);
    while (__hip_atomic_load(c, __ATOMIC_ACQUIRE, __HIP_MEMORY_SCOPE_AGENT) < target) {
      __builtin_amdgcn_s_sleep(1);
    }
  }
  __syncthreads();
  __threadfence();            // acquire side for all waves
}

// ------------------------------------------------------------ A fragment ----
// A 16x32 bf16 k order used here (same as B): lane<16: elems0-15 = K k0..k0+15
//                      lane>=16: K k0+16..k0+31. Rows contiguous in K.
__device__ inline v16bf load_frag_strided(const bf16_t* __restrict__ base) {
  const uv4* q = (const uv4*)base;
  U8x32 s;
  s.a = q[0];          // K block +0
  s.b = q[1];          // K block +8 bf16
  return __builtin_bit_cast(v16bf, s);
}

// ---------------------------------------------------------- persistent GRU --
__global__ __launch_bounds__(256, 1)
void gru_step_kernel(const bf16_t* __restrict__ xe,
                     const float*  __restrict__ bz,
                     const float*  __restrict__ br,
                     const float*  __restrict__ bh,
                     const bf16_t* __restrict__ Wswz,
                     bf16_t* __restrict__ hxh,
                     bf16_t* __restrict__ rh,
                     float*  __restrict__ out,
                     unsigned* __restrict__ cnt) {
  __shared__ uv4 ldsW[9216];                 // 144 KB: [gate(3)][nt(2)][c(24)][lane(32)][2 uv4]

  const int wg    = blockIdx.x;
  const int tid   = threadIdx.x;
  const int lane  = tid & 31;
  const int wave  = tid >> 5;
  const int mtile = wave >> 1;               // 4 row tiles of the 64-batch
  const int ntile = wave & 1;                // 2 col tiles of this WG's 32 cols
  const int hi8   = (lane >= 16) ? 8 : 0;
  const int m0    = mtile * 16;
  const int lc    = lane & 15;
  const int col   = wg * 32 + ntile * 16 + lc;
  const int arow  = m0 + lc;                 // this lane's A-matrix row
  const int off0  = (m0 + hi8) * HD + col;   // base index for C-tile element j=0

  // ---- stage this WG's weight slices (z,r,h x 2 colTiles x 24 chunks) ----
  const uv4* Wg = (const uv4*)Wswz;
  for (int u = tid; u < 9216; u += 256) {
    int f  = u >> 6;                          // local fragment 0..143
    int w  = u & 63;
    int c  = f % KCH;
    int gn = f / KCH;                         // 0..5 -> (gate, nt)
    int g  = gn >> 1, nt = gn & 1;
    int gf = (g * 32 + wg * 2 + nt) * KCH + c;
    ldsW[u] = Wg[(size_t)gf * 64 + w];
  }
  __syncthreads();

  const float bzc = bz[col], brc = br[col], bhc = bh[col];

  // h carried in registers: this wave's C-tile positions are exactly the
  // h values it computes each step, so no reload from memory is ever needed.
  float hcur[8];
#pragma unroll
  for (int j = 0; j < 8; ++j) hcur[j] = 0.0f;

  const bf16_t* hbase = hxh + arow * HD + hi8 + hi8;   // h-part A row base
  const bf16_t* rbase = rh  + arow * HD + hi8 + hi8;   // r*h-part A row base

  for (int t = 0; t < L_SEQ; ++t) {
    const bf16_t* ebase = xe + (size_t)t * (NB * ED) + arow * ED + hi8 + hi8;

    // warm L2 with next step's embedding row for this lane
    if (t + 1 < L_SEQ)
      __builtin_prefetch(ebase + (NB * ED), 0, 1);

    // ---- load the 8 step-constant embedding A-fragments once (registers) --
    v16bf ea[8];
#pragma unroll
    for (int c = 0; c < 8; ++c) ea[c] = load_frag_strided(ebase + c * 32);

    // ---------------- phase A: z and r gates (shared A fragments) ----------
    v8f accz = {0.f,0.f,0.f,0.f,0.f,0.f,0.f,0.f};
    v8f accr = {0.f,0.f,0.f,0.f,0.f,0.f,0.f,0.f};
#pragma unroll 4
    for (int c = 0; c < 16; ++c) {            // h-part chunks
      v16bf a = load_frag_strided(hbase + c * 32);
      int bzi = ((0 * 2 + ntile) * KCH + c) * 64 + lane * 2;
      int bri = ((1 * 2 + ntile) * KCH + c) * 64 + lane * 2;
      U8x32 sz{ldsW[bzi], ldsW[bzi + 1]};
      U8x32 sr{ldsW[bri], ldsW[bri + 1]};
      accz = __builtin_amdgcn_wmma_f32_16x16x32_bf16(false, a, false, __builtin_bit_cast(v16bf, sz), (short)0, accz, false, false);
      accr = __builtin_amdgcn_wmma_f32_16x16x32_bf16(false, a, false, __builtin_bit_cast(v16bf, sr), (short)0, accr, false, false);
    }
#pragma unroll
    for (int c = 0; c < 8; ++c) {             // e-part chunks from registers
      int bzi = ((0 * 2 + ntile) * KCH + 16 + c) * 64 + lane * 2;
      int bri = ((1 * 2 + ntile) * KCH + 16 + c) * 64 + lane * 2;
      U8x32 sz{ldsW[bzi], ldsW[bzi + 1]};
      U8x32 sr{ldsW[bri], ldsW[bri + 1]};
      accz = __builtin_amdgcn_wmma_f32_16x16x32_bf16(false, ea[c], false, __builtin_bit_cast(v16bf, sz), (short)0, accz, false, false);
      accr = __builtin_amdgcn_wmma_f32_16x16x32_bf16(false, ea[c], false, __builtin_bit_cast(v16bf, sr), (short)0, accr, false, false);
    }

    float zz[8];
#pragma unroll
    for (int j = 0; j < 8; ++j) {
      float zv = fast_sigmoid(accz[j] + bzc);
      float rv = fast_sigmoid(accr[j] + brc);
      zz[j] = zv;
      rh[off0 + j * HD] = (bf16_t)(rv * hcur[j]);         // publish r*h (bf16)
    }

    grid_barrier(&cnt[2 * t], GWG);

    // ---------------- phase B: candidate h~ and gate -----------------------
    // e-part first: operands already in registers, WMMA starts immediately
    // after the barrier while the rh fragment loads stream in.
    v8f acch = {0.f,0.f,0.f,0.f,0.f,0.f,0.f,0.f};
#pragma unroll
    for (int c = 0; c < 8; ++c) {
      int bhi = ((2 * 2 + ntile) * KCH + 16 + c) * 64 + lane * 2;
      U8x32 sh{ldsW[bhi], ldsW[bhi + 1]};
      acch = __builtin_amdgcn_wmma_f32_16x16x32_bf16(false, ea[c], false, __builtin_bit_cast(v16bf, sh), (short)0, acch, false, false);
    }
#pragma unroll 4
    for (int c = 0; c < 16; ++c) {            // r*h-part chunks
      v16bf a = load_frag_strided(rbase + c * 32);
      int bhi = ((2 * 2 + ntile) * KCH + c) * 64 + lane * 2;
      U8x32 sh{ldsW[bhi], ldsW[bhi + 1]};
      acch = __builtin_amdgcn_wmma_f32_16x16x32_bf16(false, a, false, __builtin_bit_cast(v16bf, sh), (short)0, acch, false, false);
    }

    float* outt = out + (size_t)t * (NB * HD);
#pragma unroll
    for (int j = 0; j < 8; ++j) {
      float ht = fast_tanh(acch[j] + bhc);
      float hn = hcur[j] + zz[j] * (ht - hcur[j]);        // (1-z)h + z*h~
      hcur[j] = hn;                                       // carry for step t+1
      outt[off0 + j * HD] = hn;                           // hs[t] (fp32)
      hxh[off0 + j * HD]  = (bf16_t)hn;                   // next step's A input
    }

    grid_barrier(&cnt[2 * t + 1], GWG);
  }
}

// ---------------------------------------------------------------- launch ----
extern "C" void kernel_launch(void* const* d_in, const int* in_sizes, int n_in,
                              void* d_out, int out_size, void* d_ws, size_t ws_size,
                              hipStream_t stream) {
  const int*   x   = (const int*)  d_in[0];
  const float* emb = (const float*)d_in[1];
  const float* Wz  = (const float*)d_in[2];
  const float* bz  = (const float*)d_in[3];
  const float* Wr  = (const float*)d_in[4];
  const float* br  = (const float*)d_in[5];
  const float* Wh  = (const float*)d_in[6];
  const float* bh  = (const float*)d_in[7];
  float* out = (float*)d_out;

  char* ws = (char*)d_ws;
  unsigned* cnt  = (unsigned*)(ws + WS_CNT);
  bf16_t*   hxh  = (bf16_t*)  (ws + WS_HXH);
  bf16_t*   rhb  = (bf16_t*)  (ws + WS_RH);
  bf16_t*   Wswz = (bf16_t*)  (ws + WS_WSWZ);
  bf16_t*   xe   = (bf16_t*)  (ws + WS_XE);

  gru_init<<<64, 256, 0, stream>>>(cnt, (unsigned*)hxh);
  gru_embed<<<4096, 256, 0, stream>>>(x, emb, xe);            // 512*64*256/8 threads
  gru_swizzle<<<288, 256, 0, stream>>>(Wz, Wr, Wh, Wswz);     // 73728 threads
  gru_step_kernel<<<GWG, 256, 0, stream>>>(xe, bz, br, bh, Wswz, hxh, rhb, out, cnt);
}
